// BasicBlock_CSAFR_51702816309692
// MI455X (gfx1250) — compile-verified, matrix operand order checked by reading, not run on hardware
//
#include <hip/hip_runtime.h>

// ---------------- problem constants ----------------
static constexpr int NB   = 64;          // batch
static constexpr int CH   = 512;         // channels
static constexpr int HH   = 32;
static constexpr int WW   = 32;
static constexpr int HP   = 34;          // padded (halo) dims
static constexpr int WP   = 34;
static constexpr int HID  = 240;
static constexpr int NCLS = 10;
static constexpr int HWSZ = HH * WW;                                  // 1024
static constexpr long long NCHW_TOT = (long long)NB * CH * HWSZ;      // 33,554,432
static constexpr long long PAD_TOT  = (long long)NB * HP * WP * CH;   // 37,879,808

#if __has_builtin(__builtin_amdgcn_global_load_async_to_lds_b128)
#define HAVE_ASYNC_LDS 1
#endif

// ---------------- vector types ----------------
typedef __attribute__((ext_vector_type(16))) __bf16 v16bf;
typedef __attribute__((ext_vector_type(8)))  float  v8f;
typedef __attribute__((ext_vector_type(4)))  float  f32x4;
typedef __attribute__((ext_vector_type(4)))  int    v4i;

typedef __attribute__((address_space(1))) v4i* gptr_v4i;
typedef __attribute__((address_space(3))) v4i* lptr_v4i;

union Frag32B { v16bf v; f32x4 f4[2]; };     // 32-byte bf16 WMMA fragment
union Pack8   { f32x4 f4; unsigned short us[8]; };

// ---------------- bf16 helpers ----------------
static __device__ __forceinline__ float bf2f(unsigned short u) {
    union { unsigned int i; float f; } v; v.i = ((unsigned int)u) << 16; return v.f;
}
static __device__ __forceinline__ unsigned short f2bf(float f) {
    union { float f; unsigned int i; } v; v.f = f;
    unsigned int i = v.i;
    unsigned int r = (i + 0x7FFFu + ((i >> 16) & 1u)) >> 16;  // RNE
    return (unsigned short)r;
}

// =====================================================================
// K0: pack OIHW f32 conv weights into bf16 WMMA B-fragment layout.
// idx = (((tap*32 + coTile)*16 + kChunk)*32 + lane)*16 + j
// =====================================================================
__global__ void pack_weights_k(const float* __restrict__ w, unsigned short* __restrict__ wp) {
    int idx = blockIdx.x * 256 + threadIdx.x;
    const int TOT = 9 * CH * CH;
    if (idx >= TOT) return;
    int j    =  idx        & 15;
    int lane = (idx >> 4)  & 31;
    int kc   = (idx >> 9)  & 15;
    int rest =  idx >> 13;
    int coT  =  rest & 31;
    int tap  =  rest >> 5;
    int k  = (lane < 16) ? ((j < 8) ? j : j + 8) : ((j < 8) ? j + 8 : j + 16);
    int ci = kc * 32 + k;
    int co = coT * 16 + (lane & 15);
    wp[idx] = f2bf(w[((size_t)co * CH + ci) * 9 + tap]);
}

// =====================================================================
// K0b: zero the halo border of a padded NHWC bf16 buffer (132 border
// positions per n, each a full 512-channel run). 8 bf16 per thread.
// =====================================================================
__global__ void halo_zero_k(unsigned short* __restrict__ p) {
    int t = blockIdx.x * 256 + threadIdx.x;
    if (t >= NB * 132 * 64) return;
    int cchunk = t & 63;
    int h = (t >> 6) % 132;
    int n = t / (132 * 64);
    int yp, xp;
    if (h < 34)      { yp = 0;  xp = h; }
    else if (h < 68) { yp = 33; xp = h - 34; }
    else { int h2 = h - 68; yp = 1 + (h2 >> 1); xp = (h2 & 1) ? 33 : 0; }
    f32x4 z = {0.f, 0.f, 0.f, 0.f};
    *(f32x4*)(p + (((size_t)n * HP + yp) * WP + xp) * CH + cchunk * 8) = z;
}

// =====================================================================
// K1: out1 = relu(bn1(x)) -> padded NHWC bf16 interior; g[n][c] = mean.
// =====================================================================
__global__ void bn1_relu_mean_k(const float* __restrict__ x,
                                const float* __restrict__ g1, const float* __restrict__ b1,
                                const float* __restrict__ m1, const float* __restrict__ v1,
                                unsigned short* __restrict__ out1, float* __restrict__ gmean) {
    __shared__ float red[256];
    int b = blockIdx.x;
    int c = b & (CH - 1);
    int n = b >> 9;
    int tid = threadIdx.x;
    float inv  = g1[c] * rsqrtf(v1[c] + 1e-5f);
    float beta = b1[c] - m1[c] * inv;
    const float* xp = x + ((size_t)n * CH + c) * HWSZ;
    float sum = 0.f;
    for (int i = tid; i < HWSZ; i += 256) {
        int y = i >> 5, xx = i & 31;
        float o = fmaxf(xp[i] * inv + beta, 0.f);
        sum += o;
        out1[(((size_t)n * HP + y + 1) * WP + xx + 1) * CH + c] = f2bf(o);
    }
    red[tid] = sum; __syncthreads();
    for (int s = 128; s > 0; s >>= 1) { if (tid < s) red[tid] += red[tid + s]; __syncthreads(); }
    if (tid == 0) gmean[n * CH + c] = red[0] * (1.f / HWSZ);
}

// =====================================================================
// K2: probe MLP + analytic gradient mask. One block per batch element.
// =====================================================================
__global__ void probe_mask_k(const float* __restrict__ gmean,
                             const float* __restrict__ fc1w, const float* __restrict__ fc1b,
                             const float* __restrict__ fc2w, const float* __restrict__ fc2b,
                             float* __restrict__ mask, float* __restrict__ probe_out) {
    __shared__ float sg[CH];
    __shared__ float sz[HID];
    __shared__ float sh[HID];
    __shared__ float st[HID];
    __shared__ float slog[16];
    __shared__ float red[256];
    __shared__ int   sidx[2];
    int n = blockIdx.x, tid = threadIdx.x;

    sg[tid] = gmean[n * CH + tid];
    sg[tid + 256] = gmean[n * CH + tid + 256];
    __syncthreads();

    if (tid < HID) {
        float z = fc1b[tid];
        const float* wr = fc1w + (size_t)tid * CH;
        for (int c = 0; c < CH; ++c) z += sg[c] * wr[c];
        sz[tid] = z;
        float a = 5.f * z;
        sh[tid] = (a > 20.f) ? z : log1pf(expf(a)) * 0.2f;
    }
    __syncthreads();

    if (tid < NCLS) {
        float lg = fc2b[tid];
        const float* wr = fc2w + (size_t)tid * HID;
        for (int k = 0; k < HID; ++k) lg += sh[k] * wr[k];
        slog[tid] = lg;
        probe_out[n * NCLS + tid] = lg;
    }
    __syncthreads();

    if (tid == 0) {
        int i1 = 0;
        for (int j = 1; j < NCLS; ++j) if (slog[j] > slog[i1]) i1 = j;
        int i2 = (i1 == 0) ? 1 : 0;
        for (int j = 0; j < NCLS; ++j) if (j != i1 && slog[j] > slog[i2]) i2 = j;
        sidx[0] = i1; sidx[1] = i2;
    }
    __syncthreads();

    if (tid < HID) {
        float wsel = fc2w[(size_t)sidx[0] * HID + tid] + fc2w[(size_t)sidx[1] * HID + tid];
        float sig = 1.f / (1.f + expf(-5.f * sz[tid]));
        st[tid] = sig * wsel;
    }
    __syncthreads();

    for (int c = tid; c < CH; c += 256) {
        float s = 0.f;
        for (int k = 0; k < HID; ++k) s += st[k] * fc1w[(size_t)k * CH + c];
        sg[c] = s;
    }
    __syncthreads();

    float lm = fmaxf(sg[tid], sg[tid + 256]);
    red[tid] = lm; __syncthreads();
    for (int s = 128; s > 0; s >>= 1) { if (tid < s) red[tid] = fmaxf(red[tid], red[tid + s]); __syncthreads(); }
    float mx = red[0]; __syncthreads();
    float le = expf(sg[tid] - mx) + expf(sg[tid + 256] - mx);
    red[tid] = le; __syncthreads();
    for (int s = 128; s > 0; s >>= 1) { if (tid < s) red[tid] += red[tid + s]; __syncthreads(); }
    float inv_sum = 2.f / red[0];
    mask[n * CH + tid]       = expf(sg[tid] - mx) * inv_sum;
    mask[n * CH + tid + 256] = expf(sg[tid + 256] - mx) * inv_sum;
}

// =====================================================================
// K3: padded buffer *= mask[n][c] (halo is 0 so scaling it is harmless).
// =====================================================================
__global__ void apply_mask_k(unsigned short* __restrict__ buf, const float* __restrict__ mask) {
    long long t = (long long)blockIdx.x * 256 + threadIdx.x;
    long long e = t * 8;
    if (e >= PAD_TOT) return;
    int n = (int)(e / ((long long)HP * WP * CH));
    int c = (int)(e & (CH - 1));
    Pack8 p; p.f4 = *(const f32x4*)(buf + e);
    #pragma unroll
    for (int i = 0; i < 8; ++i)
        p.us[i] = f2bf(bf2f(p.us[i]) * mask[n * CH + c + i]);
    *(f32x4*)(buf + e) = p.f4;
}

// =====================================================================
// Conv 3x3 via implicit GEMM, v_wmma_f32_16x16x32_bf16.
// Block = 256 threads = 8 waves; block tile = (16 x) x (128 co).
// Input slab (3 rows x 18 x x 512 ci = 55 KB) staged in LDS once
// (async-to-LDS when available), then 9 taps x 16 K-chunks of WMMA
// with register double-buffering of the B (weight) fragment.
// =====================================================================
static constexpr int SLAB_ELEMS  = 3 * 18 * CH;        // 27648 bf16
static constexpr int SLAB_CHUNKS = SLAB_ELEMS / 8;     // 3456 x 16B

static __device__ __forceinline__ void stage_slab(unsigned short* slab,
                                                  const unsigned short* __restrict__ in,
                                                  int n, int y, int x0) {
    for (int i = threadIdx.x; i < SLAB_CHUNKS; i += 256) {
        int row = i / (18 * CH / 8);
        int off = i - row * (18 * CH / 8);
        const unsigned short* src =
            in + (((size_t)n * HP + y + row) * WP + x0) * CH + (size_t)off * 8;
#if HAVE_ASYNC_LDS
        v4i* srcg = (v4i*)(void*)const_cast<unsigned short*>(src);
        v4i* dstl = (v4i*)(void*)(slab + (size_t)i * 8);
        __builtin_amdgcn_global_load_async_to_lds_b128(
            (gptr_v4i)srcg, (lptr_v4i)dstl, 0, 0);
#else
        *(f32x4*)(slab + (size_t)i * 8) = *(const f32x4*)src;
#endif
    }
#if HAVE_ASYNC_LDS
#if __has_builtin(__builtin_amdgcn_s_wait_asynccnt)
    __builtin_amdgcn_s_wait_asynccnt(0);
#else
    asm volatile("s_wait_asynccnt 0x0" ::: "memory");
#endif
#endif
    __syncthreads();
}

static __device__ __forceinline__ v8f conv_tile(const unsigned short* slab,
                                                const unsigned short* __restrict__ wp,
                                                int coT, int lane, int lm, int hi) {
    v8f acc = {};
    const unsigned short* wbase = wp + (size_t)coT * 16 * CH + lane * 16;
    Frag32B bcur;
    { const f32x4* q = (const f32x4*)wbase; bcur.f4[0] = q[0]; bcur.f4[1] = q[1]; }
    #pragma unroll
    for (int tap = 0; tap < 9; ++tap) {
        int dy = tap / 3, dx = tap % 3;
        const unsigned short* abase = slab + ((dy * 18 + lm + dx) * CH) + hi * 8;
        #pragma unroll
        for (int kc = 0; kc < 16; ++kc) {
            // issue next B load one iteration ahead of its consuming WMMA
            Frag32B bnext;
            int nkc = kc + 1, ntap = tap;
            if (nkc == 16) { nkc = 0; ntap = tap + 1; }
            if (ntap < 9) {
                const f32x4* q = (const f32x4*)(wbase + ((size_t)ntap * 32 * 16 + nkc) * CH);
                bnext.f4[0] = q[0]; bnext.f4[1] = q[1];
            } else { bnext = bcur; }
            Frag32B a;                               // A from LDS (ds_load_b128 x2)
            a.f4[0] = *(const f32x4*)(abase + kc * 32);
            a.f4[1] = *(const f32x4*)(abase + kc * 32 + 16);
            acc = __builtin_amdgcn_wmma_f32_16x16x32_bf16(
                false, a.v, false, bcur.v, (short)0, acc, false, false);
            bcur = bnext;
        }
    }
    return acc;
}

__global__ void __launch_bounds__(256)
conv1_bn2_relu_k(const unsigned short* __restrict__ in,   // padded masked NHWC bf16
                 const unsigned short* __restrict__ wp,
                 const float* __restrict__ g2, const float* __restrict__ b2,
                 const float* __restrict__ m2, const float* __restrict__ v2,
                 unsigned short* __restrict__ out) {      // padded NHWC bf16
    __shared__ unsigned short slab[SLAB_ELEMS];
    int b = blockIdx.x;
    int cb = b & 3;  b >>= 2;
    int xt = b & 1;  b >>= 1;
    int y  = b & 31;
    int n  = b >> 5;
    int x0 = xt * 16;
    stage_slab(slab, in, n, y, x0);

    int wave = threadIdx.x >> 5, lane = threadIdx.x & 31;
    int lm = lane & 15, hi = lane >> 4;
    int coT = cb * 8 + wave;
    int co  = coT * 16 + lm;

    v8f acc = conv_tile(slab, wp, coT, lane, lm, hi);

    float inv  = g2[co] * rsqrtf(v2[co] + 1e-5f);
    float beta = b2[co] - m2[co] * inv;
    #pragma unroll
    for (int r = 0; r < 8; ++r) {
        int xw = x0 + r + hi * 8;                   // D: M = r + 8*(lane/16)
        float val = fmaxf(acc[r] * inv + beta, 0.f);
        out[(((size_t)n * HP + y + 1) * WP + xw + 1) * CH + co] = f2bf(val);
    }
}

__global__ void __launch_bounds__(256)
conv2_residual_k(const unsigned short* __restrict__ in,   // padded NHWC bf16
                 const unsigned short* __restrict__ wp,
                 const float* __restrict__ xin,           // NCHW f32
                 float* __restrict__ out) {               // NCHW f32
    __shared__ unsigned short slab[SLAB_ELEMS];
    int b = blockIdx.x;
    int cb = b & 3;  b >>= 2;
    int xt = b & 1;  b >>= 1;
    int y  = b & 31;
    int n  = b >> 5;
    int x0 = xt * 16;
    stage_slab(slab, in, n, y, x0);

    int wave = threadIdx.x >> 5, lane = threadIdx.x & 31;
    int lm = lane & 15, hi = lane >> 4;
    int coT = cb * 8 + wave;
    int co  = coT * 16 + lm;

    v8f acc = conv_tile(slab, wp, coT, lane, lm, hi);

    size_t base = (((size_t)n * CH + co) * HH + y) * WW + x0 + hi * 8;    // 8 consecutive x
    const f32x4* xr = (const f32x4*)(xin + base);
    f32x4* orow = (f32x4*)(out + base);
    f32x4 lo = xr[0], hh = xr[1];
    lo.x += acc[0]; lo.y += acc[1]; lo.z += acc[2]; lo.w += acc[3];
    hh.x += acc[4]; hh.y += acc[5]; hh.z += acc[6]; hh.w += acc[7];
    orow[0] = lo; orow[1] = hh;
}

// =====================================================================
// launch
// =====================================================================
extern "C" void kernel_launch(void* const* d_in, const int* in_sizes, int n_in,
                              void* d_out, int out_size, void* d_ws, size_t ws_size,
                              hipStream_t stream) {
    const float* x       = (const float*)d_in[0];
    const float* bn1_g   = (const float*)d_in[1];
    const float* bn1_b   = (const float*)d_in[2];
    const float* bn1_m   = (const float*)d_in[3];
    const float* bn1_v   = (const float*)d_in[4];
    const float* conv1_w = (const float*)d_in[5];
    const float* bn2_g   = (const float*)d_in[6];
    const float* bn2_b   = (const float*)d_in[7];
    const float* bn2_m   = (const float*)d_in[8];
    const float* bn2_v   = (const float*)d_in[9];
    const float* conv2_w = (const float*)d_in[10];
    const float* fc1_w   = (const float*)d_in[11];
    const float* fc1_b   = (const float*)d_in[12];
    const float* fc2_w   = (const float*)d_in[13];
    const float* fc2_b   = (const float*)d_in[14];
    float* out = (float*)d_out;

    // workspace carve-up
    char* ws = (char*)d_ws;
    const size_t actB = (size_t)PAD_TOT * 2;                 // 75.8 MB padded bf16 NHWC
    const size_t wpB  = (size_t)9 * CH * CH * 2;             // 4.5 MB bf16
    unsigned short* maskedbuf = (unsigned short*)(ws);
    unsigned short* hbuf      = (unsigned short*)(ws + actB);
    float*          gmean     = (float*)(ws + 2 * actB);
    float*          maskb     = (float*)(ws + 2 * actB + (size_t)NB * CH * 4);
    unsigned short* wpack1    = (unsigned short*)(ws + 2 * actB + (size_t)2 * NB * CH * 4);
    unsigned short* wpack2    = (unsigned short*)((char*)wpack1 + wpB);

    const int packBlocks = (9 * CH * CH + 255) / 256;        // 9216
    pack_weights_k<<<packBlocks, 256, 0, stream>>>(conv1_w, wpack1);
    pack_weights_k<<<packBlocks, 256, 0, stream>>>(conv2_w, wpack2);

    const int haloBlocks = (NB * 132 * 64 + 255) / 256;      // 2112
    halo_zero_k<<<haloBlocks, 256, 0, stream>>>(maskedbuf);
    halo_zero_k<<<haloBlocks, 256, 0, stream>>>(hbuf);

    bn1_relu_mean_k<<<NB * CH, 256, 0, stream>>>(x, bn1_g, bn1_b, bn1_m, bn1_v,
                                                 maskedbuf, gmean);

    probe_mask_k<<<NB, 256, 0, stream>>>(gmean, fc1_w, fc1_b, fc2_w, fc2_b,
                                         maskb, out + NCHW_TOT);

    apply_mask_k<<<(int)((PAD_TOT / 8 + 255) / 256), 256, 0, stream>>>(maskedbuf, maskb);

    const int convBlocks = NB * HH * (WW / 16) * (CH / 128); // 16384
    conv1_bn2_relu_k<<<convBlocks, 256, 0, stream>>>(maskedbuf, wpack1,
                                                     bn2_g, bn2_b, bn2_m, bn2_v, hbuf);
    conv2_residual_k<<<convBlocks, 256, 0, stream>>>(hbuf, wpack2, x, out);
}
